// ModulatedConv2d_29480655520446
// MI455X (gfx1250) — matrix operand order repaired and checked by reading, not compiled, not run on hardware
//
#include <hip/hip_runtime.h>
#include <stdint.h>

typedef __attribute__((ext_vector_type(16))) __bf16 v16bf;
typedef __attribute__((ext_vector_type(8)))  __bf16 v8bf;
typedef __attribute__((ext_vector_type(8)))  float  v8f;

#define B_    16
#define IC_   256
#define OC_   256
#define DL_   512
#define H_    64
#define W_    64
#define KK_   9
#define MOD_SCALE 0.0625f                 /* sqrt(2/512) exactly */
#define W_SCALE   0.029462782549439483f   /* sqrt(2/(256*9))     */

/* ------------------------------------------------------------------ */
/* 1) s[b][ic] = sum_d style[b][d] * mod_weight[ic][d] * MOD_SCALE    */
/* ------------------------------------------------------------------ */
__global__ void k_style(const float* __restrict__ style,
                        const float* __restrict__ modw,
                        float* __restrict__ sArr) {
  int idx = blockIdx.x * blockDim.x + threadIdx.x;   // 4096 = B*IC
  int b = idx >> 8, ic = idx & 255;
  const float* st = style + (size_t)b * DL_;
  const float* mw = modw + (size_t)ic * DL_;
  float acc = 0.f;
#pragma unroll 8
  for (int d = 0; d < DL_; ++d) acc = fmaf(st[d], mw[d], acc);
  sArr[idx] = acc * MOD_SCALE;
}

/* ------------------------------------------------------------------ */
/* 2) demod[oc] = rsqrt(sum_{ic,kh,kw} w^2 + eps)  (UNSCALED weight)  */
/* ------------------------------------------------------------------ */
__global__ void k_demod(const float* __restrict__ weight,
                        float* __restrict__ demod) {
  int oc = threadIdx.x;                              // 256 threads
  const float* w = weight + (size_t)oc * (IC_ * KK_);
  float acc = 0.f;
#pragma unroll 8
  for (int i = 0; i < IC_ * KK_; ++i) acc = fmaf(w[i], w[i], acc);
  demod[oc] = rsqrtf(acc + 1e-8f);
}

/* ------------------------------------------------------------------ */
/* 3) repack weight -> bf16, fold w_scale*demod.                      */
/*    wq layout: [icc(8)][ocb(4)][tap(9)][ocl(64)][ici(32)]           */
/*    => per-(block,K-chunk) slab is one CONTIGUOUS 36 KB region.     */
/* ------------------------------------------------------------------ */
__global__ void k_wq(const float* __restrict__ weight,
                     const float* __restrict__ demod,
                     unsigned short* __restrict__ wq) {
  int idx = blockIdx.x * blockDim.x + threadIdx.x;   // 589824
  int ici = idx & 31;
  int t = idx >> 5;
  int ocl = t & 63; t >>= 6;
  int tap = t % 9;  t /= 9;
  int ocb = t & 3;  t >>= 2;
  int icc = t;                                       // 0..7
  int oc = ocb * 64 + ocl;
  int ic = icc * 32 + ici;
  float v = weight[((size_t)oc * IC_ + ic) * KK_ + tap] * W_SCALE * demod[oc];
  __bf16 hv = (__bf16)v;
  wq[idx] = __builtin_bit_cast(unsigned short, hv);
}

/* ------------------------------------------------------------------ */
/* 4) implicit-GEMM conv.  Block = (b, h row, 64-OC band).            */
/*    M=64 (oc) x N=64 (w) tile, K = 256 ic * 9 taps.                 */
/*    128 threads = 4 waves in a 2x2 (M,N) arrangement; each wave     */
/*    owns 2x2 WMMA 16x16 tiles (v8f accumulators).                   */
/* ------------------------------------------------------------------ */
#define W_TILE_ELEMS (9 * 64 * 32)   /* 18432 bf16 = 36864 B (async slab) */
#define X_STRIDE     40              /* 32 ic padded to 40 (80B, 16B-aligned) */
#define X_TILE_ELEMS (3 * 66 * X_STRIDE)

__global__ void __launch_bounds__(128)
k_conv(const float* __restrict__ image,
       const float* __restrict__ sArr,
       const unsigned short* __restrict__ wq,
       float* __restrict__ out) {
  __shared__ __bf16 wTile[W_TILE_ELEMS];   /* [tap][ocl 64][ici 32]      */
  __shared__ __bf16 xTile[X_TILE_ELEMS];   /* [r 3][col 66][ici 32 pad40]*/

  const int tid   = threadIdx.x;
  const int lane  = tid & 31;
  const int wave  = tid >> 5;
  const int wm    = wave & 1;      /* oc half of 64-band: +32 */
  const int wn    = wave >> 1;     /* pixel half: +32          */
  const int l15   = lane & 15;
  const int lhalf = lane >> 4;

  int bid = blockIdx.x;
  const int ocb = bid & 3;  bid >>= 2;
  const int h   = bid & 63; bid >>= 6;
  const int b   = bid;

  const float* imgB = image + (size_t)b * IC_ * H_ * W_;
  const float* sB   = sArr + (size_t)b * IC_;

  /* LDS byte offset of wTile for the async DMA destination VGPR */
  const uint32_t ldsW = (uint32_t)(uintptr_t)(&wTile[0]);

  const v8f vzero = {0.f, 0.f, 0.f, 0.f, 0.f, 0.f, 0.f, 0.f};
  v8f acc[2][2] = {{vzero, vzero}, {vzero, vzero}};

  for (int icc = 0; icc < 8; ++icc) {
    /* ---- async-DMA the contiguous 36 KB weight slab into LDS ---- */
    const unsigned short* slab =
        wq + (size_t)(icc * 4 + ocb) * W_TILE_ELEMS;
    {
      uint32_t ldsDst = ldsW + (uint32_t)tid * 16u;
      uint32_t gOff   = (uint32_t)tid * 16u;
#pragma unroll
      for (int j = 0; j < 18; ++j) {   /* 18 * 128 lanes * 16B = 36864B */
        asm volatile("global_load_async_to_lds_b128 %0, %1, %2"
                     :: "v"(ldsDst + (uint32_t)j * 2048u),
                        "v"(gOff + (uint32_t)j * 2048u),
                        "s"(slab)
                     : "memory");
      }
    }

    /* ---- stage input halo tile: fp32 load * s[b,ic] -> bf16 LDS ---- */
    const int icBase = icc * 32;
    for (int li = tid; li < 3 * 66 * 32; li += 128) {
      int col = li % 66;              /* coalesced over w           */
      int t2  = li / 66;
      int ici = t2 & 31;
      int r   = t2 >> 5;
      int ic  = icBase + ici;
      int hin = h - 1 + r;
      int win = col - 1;
      float v = 0.f;
      if ((unsigned)hin < (unsigned)H_ && (unsigned)win < (unsigned)W_)
        v = imgB[((size_t)ic * H_ + hin) * W_ + win] * sB[ic];
      xTile[(r * 66 + col) * X_STRIDE + ici] = (__bf16)v;
    }

    if (icc + 1 < 8) {  /* warm L2/WGP$ for the next chunk */
      const float* nf = imgB + ((size_t)(icBase + 32) * H_ + h) * W_ + (tid & 63);
      __builtin_prefetch(nf, 0, 1);
    }

    asm volatile("s_wait_asynccnt 0x0" ::: "memory");
    __syncthreads();

    /* ---- 9 taps x 2x2 WMMA tiles ---- */
#pragma unroll
    for (int tap = 0; tap < 9; ++tap) {
      const int kh = tap / 3, kw = tap % 3;
      v16bf aF[2], bF[2];
#pragma unroll
      for (int mi = 0; mi < 2; ++mi) {
        /* A frag (this code's k order, same as B): lane half holds 16 contiguous K = half*16..+16 */
        const __bf16* base = &wTile[((tap * 64) + wm * 32 + mi * 16 + l15) * 32];
        v8bf lo = *(const v8bf*)(base + lhalf * 16);
        v8bf hi = *(const v8bf*)(base + lhalf * 16 + 8);
        aF[mi] = __builtin_shufflevector(lo, hi,
                   0, 1, 2, 3, 4, 5, 6, 7, 8, 9, 10, 11, 12, 13, 14, 15);
      }
#pragma unroll
      for (int ni = 0; ni < 2; ++ni) {
        /* B layout: lane = column n, 16 contiguous K per lane half */
        const int px = wn * 32 + ni * 16 + l15;
        bF[ni] = *(const v16bf*)&xTile[(kh * 66 + px + kw) * X_STRIDE + lhalf * 16];
      }
#pragma unroll
      for (int mi = 0; mi < 2; ++mi)
#pragma unroll
        for (int ni = 0; ni < 2; ++ni)
          acc[mi][ni] = __builtin_amdgcn_wmma_f32_16x16x32_bf16(
              false, aF[mi], false, bF[ni], (short)0, acc[mi][ni],
              false, false);
    }
    __syncthreads();
  }

  /* ---- store: C layout VGPR r -> (M = r + half*8, N = l15) ---- */
  float* outB = out + (((size_t)b * OC_ + ocb * 64) * H_ + h) * W_;
#pragma unroll
  for (int mi = 0; mi < 2; ++mi) {
#pragma unroll
    for (int ni = 0; ni < 2; ++ni) {
      const int w    = wn * 32 + ni * 16 + l15;
      const int ocl0 = wm * 32 + mi * 16 + lhalf * 8;
#pragma unroll
      for (int r = 0; r < 8; ++r)
        outB[(size_t)(ocl0 + r) * (H_ * W_) + w] = acc[mi][ni][r];
    }
  }
}

/* ------------------------------------------------------------------ */
extern "C" void kernel_launch(void* const* d_in, const int* in_sizes, int n_in,
                              void* d_out, int out_size, void* d_ws, size_t ws_size,
                              hipStream_t stream) {
  const float* image  = (const float*)d_in[0];
  const float* style  = (const float*)d_in[1];
  const float* weight = (const float*)d_in[2];
  const float* modw   = (const float*)d_in[3];
  float* out = (float*)d_out;

  /* workspace: s[4096] f32 | demod[256] f32 | wq bf16 (1.18 MB) */
  float* sArr  = (float*)d_ws;
  float* demod = sArr + B_ * IC_;
  unsigned short* wq = (unsigned short*)((char*)d_ws + 17408);

  k_style<<<dim3(16),   dim3(256), 0, stream>>>(style, modw, sArr);
  k_demod<<<dim3(1),    dim3(256), 0, stream>>>(weight, demod);
  k_wq   <<<dim3(2304), dim3(256), 0, stream>>>(weight, demod, wq);
  k_conv <<<dim3(B_ * H_ * (OC_ / 64)), dim3(128), 0, stream>>>(image, sArr, wq, out);
}
